// HGRN_Module_71914932404651
// MI455X (gfx1250) — matrix operand order repaired and checked by reading, not compiled, not run on hardware
//
#include <hip/hip_runtime.h>
#include <math.h>

typedef __bf16 bf16;
typedef __attribute__((ext_vector_type(16))) __bf16 v16bf;
typedef __attribute__((ext_vector_type(8)))  float  v8f;
typedef __attribute__((ext_vector_type(4)))  int    i32x4;

#if defined(__AMDGCN__) && __has_builtin(__builtin_amdgcn_global_load_async_to_lds_b128) && \
    __has_builtin(__builtin_amdgcn_s_wait_asynccnt)
#define USE_ASYNC_LDS 1
#else
#define USE_ASYNC_LDS 0
#endif

namespace {
constexpr int K = 4, N = 16384, M = 2048, F = 256, F2 = 256, G = 8, DK = 64, DM = 32;

// workspace layout (bytes)
constexpr size_t off_Ibf = 0;                                   // K*N*M bf16  (256 MB)
constexpr size_t off_Hp  = off_Ibf + (size_t)K * N * M * 2;     // K*N*F2 bf16 (32 MB)
constexpr size_t off_He  = off_Hp  + (size_t)K * N * F2 * 2;    // K*M*F2 bf16 (4 MB)
constexpr size_t off_nv  = off_He  + (size_t)K * M * F2 * 2;    // K*N f32
constexpr size_t off_de  = off_nv  + (size_t)K * N * 4;         // K*M f32
constexpr size_t off_g   = off_de  + (size_t)K * M * 4;         // K*F f32
constexpr size_t off_wsc = off_g   + (size_t)K * F * 4;         // K*M f32
constexpr size_t off_rl  = off_wsc + (size_t)K * M * 4;         // K f32
} // namespace

// 16-byte global(bf16) -> LDS copy; async (GLOBAL_LOAD_ASYNC_TO_LDS_B128) on gfx1250.
__device__ __forceinline__ void cp16(bf16* lds_dst, const bf16* gsrc) {
#if USE_ASYNC_LDS
  __builtin_amdgcn_global_load_async_to_lds_b128(
      (__attribute__((address_space(1))) i32x4*)gsrc,
      (__attribute__((address_space(3))) i32x4*)lds_dst, 0, 0);
#else
  *(uint4*)lds_dst = *(const uint4*)gsrc;
#endif
}

__device__ __forceinline__ void cp_wait() {
#if USE_ASYNC_LDS
  __builtin_amdgcn_s_wait_asynccnt(0);
#endif
}

// ---------------- WMMA fragment helpers (CDNA5 16x16x32 bf16 layouts) ----------------
// A (16x32, MxK): lanes 0-15 -> M=lane, lanes 16-31 -> M=lane-16 with K offset +8;
// VGPR v<4 holds K pairs {h*8+2v}, VGPR v>=4 holds {16+h*8+2(v-4)}.
__device__ __forceinline__ v16bf frag_a_rows(const bf16* p, int stride, int lane) {
  int m = lane & 15, h = lane >> 4;
  const bf16* r = p + m * stride;
  v16bf a;
#pragma unroll
  for (int v = 0; v < 4; ++v) {
    int kk = h * 8 + 2 * v;
    a[2 * v]     = r[kk];
    a[2 * v + 1] = r[kk + 1];
  }
#pragma unroll
  for (int v = 0; v < 4; ++v) {
    int kk = 16 + h * 8 + 2 * v;
    a[8 + 2 * v]     = r[kk];
    a[8 + 2 * v + 1] = r[kk + 1];
  }
  return a;
}

// A fragment, LDS tile stored [k][m] (transposed source, e.g. I^T GEMM).
__device__ __forceinline__ v16bf frag_a_cols(const bf16* p, int stride, int lane) {
  int m = lane & 15, h = lane >> 4;
  v16bf a;
#pragma unroll
  for (int v = 0; v < 4; ++v) {
    int kk = h * 8 + 2 * v;
    a[2 * v]     = p[kk * stride + m];
    a[2 * v + 1] = p[(kk + 1) * stride + m];
  }
#pragma unroll
  for (int v = 0; v < 4; ++v) {
    int kk = 16 + h * 8 + 2 * v;
    a[8 + 2 * v]     = p[kk * stride + m];
    a[8 + 2 * v + 1] = p[(kk + 1) * stride + m];
  }
  return a;
}

// B (32x16, KxN): lanes 0-15 -> N=lane, lanes 16-31 -> N=lane-16; K order same as A (h*8+s, 16+h*8+(s-8)).
// LDS tile stored [n][k].
__device__ __forceinline__ v16bf frag_b_rows(const bf16* p, int stride, int lane) {
  int n = lane & 15, h = lane >> 4;
  const bf16* r = p + n * stride;
  v16bf b;
#pragma unroll
  for (int v = 0; v < 8; ++v) {
    int kk = h * 8 + 2 * v + (v >= 4 ? 8 : 0);
    b[2 * v]     = r[kk];
    b[2 * v + 1] = r[kk + 1];
  }
  return b;
}

// B fragment, LDS tile stored natural [k][n].
__device__ __forceinline__ v16bf frag_b_cols(const bf16* p, int stride, int lane) {
  int n = lane & 15, h = lane >> 4;
  v16bf b;
#pragma unroll
  for (int v = 0; v < 8; ++v) {
    int kk = h * 8 + 2 * v + (v >= 4 ? 8 : 0);
    b[2 * v]     = p[kk * stride + n];
    b[2 * v + 1] = p[(kk + 1) * stride + n];
  }
  return b;
}

__device__ __forceinline__ v8f wmma_bf16(v16bf a, v16bf b, v8f c) {
  return __builtin_amdgcn_wmma_f32_16x16x32_bf16(false, a, false, b, (short)0, c, false, false);
}

// ---------------- Kernel 0: zero d_e accumulator ----------------
__global__ void k_zero_de(float* __restrict__ de) {
  int i = blockIdx.x * blockDim.x + threadIdx.x;
  if (i < K * M) de[i] = 0.f;
}

// ---------------- Kernel 1: single pass over fp32 incidence ----------------
// row sums -> norm_v, col sums -> d_e (atomic), fp32 -> bf16 conversion.
__global__ void k_prep(const float* __restrict__ I, bf16* __restrict__ Ib,
                       float* __restrict__ nv, float* __restrict__ de) {
  int k = blockIdx.y, nb = blockIdx.x * 64, tid = threadIdx.x;
  __shared__ float rowsum[64];
  if (tid < 64) rowsum[tid] = 0.f;
  __syncthreads();
  int col0 = tid * 8;  // 256 threads * 8 cols = 2048 = M
  float colacc[8] = {0, 0, 0, 0, 0, 0, 0, 0};
  for (int r = 0; r < 64; ++r) {
    size_t base = ((size_t)k * N + nb + r) * M + col0;
    const float* src = I + base;
    float rs = 0.f;
#pragma unroll
    for (int j = 0; j < 8; ++j) {
      float v = src[j];
      colacc[j] += v;
      rs += v;
      Ib[base + j] = (bf16)v;
    }
    atomicAdd(&rowsum[r], rs);
  }
  __syncthreads();
  if (tid < 64) nv[(size_t)k * N + nb + tid] = rsqrtf(rowsum[tid] + 1e-6f);
#pragma unroll
  for (int j = 0; j < 8; ++j) atomicAdd(&de[k * M + col0 + j], colacc[j]);
}

// ---------------- Kernel 2: g_k = mean over N of features ----------------
__global__ void k_gmean(const float* __restrict__ feat, float* __restrict__ g) {
  int k = blockIdx.y, f = blockIdx.x, tid = threadIdx.x;
  __shared__ float red[256];
  float s = 0.f;
  for (int n = tid; n < N; n += 256) s += feat[((size_t)k * N + n) * F + f];
  red[tid] = s;
  __syncthreads();
  for (int st = 128; st > 0; st >>= 1) {
    if (tid < st) red[tid] += red[tid + st];
    __syncthreads();
  }
  if (tid == 0) g[k * F + f] = red[0] * (1.f / N);
}

// ---------------- Kernel 3: MLP routing, softmax, rl_k, w_k / d_e ----------------
__global__ void k_route(const float* __restrict__ g, const float* __restrict__ W1,
                        const float* __restrict__ b1, const float* __restrict__ W2,
                        const float* __restrict__ b2, const float* __restrict__ experts,
                        const float* __restrict__ wproj, const float* __restrict__ de,
                        float* __restrict__ wsc, float* __restrict__ rl) {
  int k = blockIdx.x, tid = threadIdx.x;
  __shared__ float gs[F];
  __shared__ float h1[DM];
  __shared__ float q[DK];
  __shared__ float kexp[G * DK];
  __shared__ float alpha[G];
  __shared__ float sc[G];
  gs[tid] = g[k * F + tid];
  __syncthreads();
  if (tid < DM) {
    float s = b1[tid];
    for (int f = 0; f < F; ++f) s += gs[f] * W1[tid * F + f];
    h1[tid] = tanhf(s);
  }
  __syncthreads();
  if (tid < DK) {
    float s = b2[tid];
    for (int d = 0; d < DM; ++d) s += h1[d] * W2[tid * DM + d];
    q[tid] = s;
  }
  __syncthreads();
  for (int idx = tid; idx < G * DK; idx += 256) {
    int gi = idx / DK, d = idx % DK;
    float s = 0.f;
    for (int m = 0; m < M; ++m) s += experts[(size_t)gi * M + m] * wproj[(size_t)d * M + m];
    kexp[idx] = s;
  }
  __syncthreads();
  if (tid < G) {
    float s = 0.f;
    for (int d = 0; d < DK; ++d) s += q[d] * kexp[tid * DK + d];
    sc[tid] = s * 0.125f;  // 1/sqrt(dk), dk=64
  }
  __syncthreads();
  if (tid == 0) {
    float mx = sc[0];
    for (int i = 1; i < G; ++i) mx = fmaxf(mx, sc[i]);
    float sum = 0.f;
    for (int i = 0; i < G; ++i) { alpha[i] = __expf(sc[i] - mx); sum += alpha[i]; }
    float inv = 1.f / sum;
    float mean = 1.f / G;  // softmax sums to 1
    float var = 0.f;
    for (int i = 0; i < G; ++i) {
      alpha[i] *= inv;
      float d = alpha[i] - mean;
      var += d * d;
    }
    var /= (G - 1);  // unbiased
    rl[k] = var / (mean * mean + 1e-8f);
  }
  __syncthreads();
  for (int m = tid; m < M; m += 256) {
    float w = 0.f;
    for (int gi = 0; gi < G; ++gi) w += alpha[gi] * experts[(size_t)gi * M + m];
    wsc[k * M + m] = w / (de[k * M + m] + 1e-6f);
  }
}

// ---------------- Kernel 4 (WMMA): Hp = (H @ theta^T) * norm_v -> bf16 ----------------
// block: 256 thr = 8 waves; output tile 64 rows x 32 cols; wave grid 4x2; K-step 64.
__global__ void k_proj(const float* __restrict__ feat, const float* __restrict__ theta,
                       const float* __restrict__ nv, bf16* __restrict__ Hp) {
  int k = blockIdx.z, nb = blockIdx.x * 64, fb = blockIdx.y * 32;
  int tid = threadIdx.x, lane = tid & 31, wave = tid >> 5;
  int wr = wave >> 1, wc = wave & 1;
  __shared__ __align__(16) bf16 At[64 * 64];  // H tile, [n][kf]
  __shared__ __align__(16) bf16 Bt[32 * 64];  // theta tile, [f2][kf]  (B(k,n)=Bt[n][k])
  v8f c = {0, 0, 0, 0, 0, 0, 0, 0};
  for (int kb = 0; kb < F; kb += 64) {
    int r = tid >> 2, c8 = (tid & 3) * 8;
#pragma unroll
    for (int half = 0; half < 2; ++half) {
      int cc = c8 + half * 32;
      const float* src = feat + ((size_t)k * N + nb + r) * F + kb + cc;
#pragma unroll
      for (int j = 0; j < 8; ++j) At[r * 64 + cc + j] = (bf16)src[j];
    }
    {
      int fr = tid >> 3, k8 = (tid & 7) * 8;
      const float* ts = theta + (size_t)(fb + fr) * F + kb + k8;
#pragma unroll
      for (int j = 0; j < 8; ++j) Bt[fr * 64 + k8 + j] = (bf16)ts[j];
    }
    __syncthreads();
    v16bf a0 = frag_a_rows(At + wr * 16 * 64, 64, lane);
    v16bf b0 = frag_b_rows(Bt + wc * 16 * 64, 64, lane);
    c = wmma_bf16(a0, b0, c);
    v16bf a1 = frag_a_rows(At + wr * 16 * 64 + 32, 64, lane);
    v16bf b1 = frag_b_rows(Bt + wc * 16 * 64 + 32, 64, lane);
    c = wmma_bf16(a1, b1, c);
    __syncthreads();
  }
  int colg = fb + wc * 16 + (lane & 15);
  int h = lane >> 4;
#pragma unroll
  for (int r8 = 0; r8 < 8; ++r8) {
    int row = nb + wr * 16 + h * 8 + r8;
    float v = c[r8] * nv[(size_t)k * N + row];
    Hp[((size_t)k * N + row) * F2 + colg] = (bf16)v;
  }
}

// ---------------- Kernel 5 (WMMA): He = (I^T @ Hp) * (w_k/d_e) -> bf16 ----------------
// K-step 64 over N; async global->LDS staging of bf16 tiles.
__global__ void k_edge(const bf16* __restrict__ Ib, const bf16* __restrict__ Hp,
                       const float* __restrict__ wsc, bf16* __restrict__ He) {
  int k = blockIdx.z, mb = blockIdx.x * 64, fb = blockIdx.y * 32;
  int tid = threadIdx.x, lane = tid & 31, wave = tid >> 5;
  int wr = wave >> 1, wc = wave & 1;
  __shared__ __align__(16) bf16 It[64 * 64];  // I tile, natural [n][m]; A(m,kn)=It[kn][m]
  __shared__ __align__(16) bf16 Bt[64 * 32];  // Hp tile, natural [n][f2]
  v8f c = {0, 0, 0, 0, 0, 0, 0, 0};
  int nrI = tid >> 3, m8 = (tid & 7) * 8;   // I tile: 2 chunks of 32 rows
  int nrB = tid >> 2, f8 = (tid & 3) * 8;   // Hp tile: 64 rows x 32 cols
  for (int nb = 0; nb < N; nb += 64) {
#pragma unroll
    for (int half = 0; half < 2; ++half) {
      int nn = nrI + half * 32;
      cp16(&It[nn * 64 + m8], Ib + ((size_t)k * N + nb + nn) * M + mb + m8);
    }
    cp16(&Bt[nrB * 32 + f8], Hp + ((size_t)k * N + nb + nrB) * F2 + fb + f8);
    if (nb + 64 < N)
      __builtin_prefetch(Ib + ((size_t)k * N + nb + 64 + nrI) * M + mb + m8, 0, 1);
    cp_wait();
    __syncthreads();
    v16bf a0 = frag_a_cols(It + wr * 16, 64, lane);
    v16bf b0 = frag_b_cols(Bt + wc * 16, 32, lane);
    c = wmma_bf16(a0, b0, c);
    v16bf a1 = frag_a_cols(It + 32 * 64 + wr * 16, 64, lane);
    v16bf b1 = frag_b_cols(Bt + 32 * 32 + wc * 16, 32, lane);
    c = wmma_bf16(a1, b1, c);
    __syncthreads();
  }
  int f2g = fb + wc * 16 + (lane & 15);
  int h = lane >> 4;
#pragma unroll
  for (int r8 = 0; r8 < 8; ++r8) {
    int m = mb + wr * 16 + h * 8 + r8;
    float v = c[r8] * wsc[k * M + m];
    He[((size_t)k * M + m) * F2 + f2g] = (bf16)v;
  }
}

// ---------------- Kernel 6 (WMMA): out = sum_k coef_k * lrelu((I @ He_k) * norm_v) ----------------
__global__ void k_node(const bf16* __restrict__ Ib, const bf16* __restrict__ He,
                       const float* __restrict__ nv, const float* __restrict__ lambdas,
                       float* __restrict__ out) {
  int nb = blockIdx.x * 64, fb = blockIdx.y * 32;
  int tid = threadIdx.x, lane = tid & 31, wave = tid >> 5;
  int wr = wave >> 1, wc = wave & 1;
  __shared__ __align__(16) bf16 It[64 * 64];  // I tile, natural [n][m]
  __shared__ __align__(16) bf16 Bt[64 * 32];  // He tile, natural [m][f2]
  float acc[8] = {0, 0, 0, 0, 0, 0, 0, 0};
  int h = lane >> 4;
  int nr = tid >> 2, c8 = (tid & 3) * 8;  // both tiles: 64 rows, 8/thread per chunk
  for (int k = 0; k < K; ++k) {
    v8f c = {0, 0, 0, 0, 0, 0, 0, 0};
    for (int mb = 0; mb < M; mb += 64) {
      const bf16* si = Ib + ((size_t)k * N + nb + nr) * M + mb + c8;
      cp16(&It[nr * 64 + c8], si);
      cp16(&It[nr * 64 + c8 + 32], si + 32);
      cp16(&Bt[nr * 32 + c8], He + ((size_t)k * M + mb + nr) * F2 + fb + c8);
      if (mb + 64 < M) __builtin_prefetch(si + 64, 0, 1);
      cp_wait();
      __syncthreads();
      v16bf a0 = frag_a_rows(It + wr * 16 * 64, 64, lane);
      v16bf b0 = frag_b_cols(Bt + wc * 16, 32, lane);
      c = wmma_bf16(a0, b0, c);
      v16bf a1 = frag_a_rows(It + wr * 16 * 64 + 32, 64, lane);
      v16bf b1 = frag_b_cols(Bt + 32 * 32 + wc * 16, 32, lane);
      c = wmma_bf16(a1, b1, c);
      __syncthreads();
    }
    float coef = (k == K - 1) ? 1.f : tanhf(lambdas[k]);
#pragma unroll
    for (int r8 = 0; r8 < 8; ++r8) {
      int row = nb + wr * 16 + h * 8 + r8;
      float v = c[r8] * nv[(size_t)k * N + row];
      v = v > 0.f ? v : 0.2f * v;  // leaky relu 0.2
      acc[r8] += coef * v;
    }
  }
  int colg = fb + wc * 16 + (lane & 15);
#pragma unroll
  for (int r8 = 0; r8 < 8; ++r8) {
    int row = nb + wr * 16 + h * 8 + r8;
    out[(size_t)row * F2 + colg] = acc[r8];
  }
}

// ---------------- Kernel 7: mean of rl_k -> last output element ----------------
__global__ void k_rl(const float* __restrict__ rl, float* __restrict__ out) {
  if (threadIdx.x == 0 && blockIdx.x == 0)
    out[(size_t)N * F2] = (rl[0] + rl[1] + rl[2] + rl[3]) * 0.25f;
}

extern "C" void kernel_launch(void* const* d_in, const int* in_sizes, int n_in,
                              void* d_out, int out_size, void* d_ws, size_t ws_size,
                              hipStream_t stream) {
  (void)in_sizes; (void)n_in; (void)out_size; (void)ws_size;
  const float* I       = (const float*)d_in[0];
  const float* feat    = (const float*)d_in[1];
  const float* W1      = (const float*)d_in[2];
  const float* b1      = (const float*)d_in[3];
  const float* W2      = (const float*)d_in[4];
  const float* b2      = (const float*)d_in[5];
  const float* experts = (const float*)d_in[6];
  const float* wproj   = (const float*)d_in[7];
  const float* theta   = (const float*)d_in[8];
  const float* lambdas = (const float*)d_in[9];
  float* out = (float*)d_out;

  char* ws = (char*)d_ws;
  bf16* Ib   = (bf16*)(ws + off_Ibf);
  bf16* Hp   = (bf16*)(ws + off_Hp);
  bf16* He   = (bf16*)(ws + off_He);
  float* nv  = (float*)(ws + off_nv);
  float* de  = (float*)(ws + off_de);
  float* g   = (float*)(ws + off_g);
  float* wsc = (float*)(ws + off_wsc);
  float* rl  = (float*)(ws + off_rl);

  k_zero_de<<<(K * M + 255) / 256, 256, 0, stream>>>(de);
  k_prep<<<dim3(N / 64, K), 256, 0, stream>>>(I, Ib, nv, de);
  k_gmean<<<dim3(F, K), 256, 0, stream>>>(feat, g);
  k_route<<<K, 256, 0, stream>>>(g, W1, b1, W2, b2, experts, wproj, de, wsc, rl);
  k_proj<<<dim3(N / 64, F2 / 32, K), 256, 0, stream>>>(feat, theta, nv, Hp);
  k_edge<<<dim3(M / 64, F2 / 32, K), 256, 0, stream>>>(Ib, Hp, wsc, He);
  k_node<<<dim3(N / 64, F2 / 32), 256, 0, stream>>>(Ib, He, nv, lambdas, out);
  k_rl<<<1, 32, 0, stream>>>(rl, out);
}
